// CrossAttentionGraphModule_67164698575218
// MI455X (gfx1250) — matrix operand order repaired and checked by reading, not compiled, not run on hardware
//
#include <hip/hip_runtime.h>

// ---------------- problem constants (fixed by setup_inputs) ----------------
#define D_DIM   512
#define LAT_DIM 768
#define B_G     16
#define S_LEN   256
#define H_N     8
#define DH_DIM  64
#define LMAX    992          // max_nodes = 512 + 32*15
#define M_NODES (B_G * LMAX) // 15872

typedef __attribute__((ext_vector_type(16))) __bf16 v16bf;
typedef __attribute__((ext_vector_type(8)))  __bf16 v8bf;
typedef __attribute__((ext_vector_type(8)))  float  v8f;

// fp32 -> bf16 round-to-nearest-even (used only in converts / epilogues)
__device__ __forceinline__ __bf16 f2bf(float f) {
    unsigned u = __builtin_bit_cast(unsigned, f);
    u += 0x7fffu + ((u >> 16) & 1u);
    unsigned short h = (unsigned short)(u >> 16);
    return __builtin_bit_cast(__bf16, h);
}

__device__ __forceinline__ v8f wmma_bf16(v16bf a, v16bf b, v8f c) {
    return __builtin_amdgcn_wmma_f32_16x16x32_bf16(false, a, false, b, (short)0, c,
                                                   false, false);
}

// A-fragment (16x32 bf16): lane holds row M=lane&15; k = e + 16*(lane>=16).
// p points at A_row + kk + 16*(lane>=16); reads p[0..15] (2x 16B loads).
__device__ __forceinline__ v16bf load_a_frag_bf(const __bf16* p) {
    v8bf lo = *(const v8bf*)(p);
    v8bf hi = *(const v8bf*)(p + 8);
    v16bf a;
#pragma unroll
    for (int e = 0; e < 8; ++e) { a[e] = lo[e]; a[8 + e] = hi[e]; }
    return a;
}

// Same A layout, bf16 source in LDS (attention probabilities).
__device__ __forceinline__ v16bf load_a_frag_lds(const __bf16* p) {
    v16bf a;
#pragma unroll
    for (int e = 0; e < 8; ++e) { a[e] = p[e]; a[8 + e] = p[8 + e]; }
    return a;
}

// B-fragment (32x16 bf16): lane holds col N=lane&15; k = e + 16*(lane>=16).
// Source layouts below guarantee the 16 elements are CONTIGUOUS at p (2x 16B loads).
__device__ __forceinline__ v16bf load_b16(const __bf16* p) {
    v8bf x0 = *(const v8bf*)(p);
    v8bf x1 = *(const v8bf*)(p + 8);
    v16bf b;
#pragma unroll
    for (int e = 0; e < 8; ++e) { b[e] = x0[e]; b[8 + e] = x1[e]; }
    return b;
}

// ---------------------------------------------------------------------------
// C[M,Nc] = A[M,K] @ W[K,Nc] + bias, A bf16 row-major, W pre-transformed to
// Wb[k/32][n][k%32] so B-fragments are contiguous.  256 thr = 8 waves; each
// wave owns a 16x32 tile (one A-frag feeds two WMMAs -> 2x arithmetic
// intensity on the A stream); a block covers 16 rows x 256 cols.  Optional
// fp32 and/or bf16 outputs (bf16 feeds the next WMMA consumer directly).
// ---------------------------------------------------------------------------
__global__ void gemm_bf16_kernel(const __bf16* __restrict__ A, const __bf16* __restrict__ Wb,
                                 const float* __restrict__ bias,
                                 float* __restrict__ Cf, __bf16* __restrict__ Cb,
                                 int M, int Nc, int K) {
    const int lane = threadIdx.x & 31;
    const int wave = threadIdx.x >> 5;
    const int hi   = lane >> 4;
    const int ln   = lane & 15;
    const int tileN = blockIdx.x * 256 + wave * 32;
    const int tileM = blockIdx.y * 16;
    if (tileN >= Nc || tileM >= M) return;

    const __bf16* arow  = A + (size_t)(tileM + ln) * K + hi * 16;
    const __bf16* bcol0 = Wb + (size_t)(tileN + ln) * 32 + hi * 16;
    const __bf16* bcol1 = Wb + (size_t)(tileN + 16 + ln) * 32 + hi * 16;
    const size_t bstep = (size_t)Nc * 32;   // advance one k/32 slab

    v8f acc0 = {}, acc1 = {};
#pragma unroll 4
    for (int kk = 0; kk < K; kk += 32) {
        __builtin_prefetch(arow + kk + 128, 0, 1);   // global_prefetch_b8
        v16bf a  = load_a_frag_bf(arow + kk);
        const size_t slab = (size_t)(kk >> 5) * bstep;
        v16bf b0 = load_b16(bcol0 + slab);
        v16bf b1 = load_b16(bcol1 + slab);
        acc0 = wmma_bf16(a, b0, acc0);
        acc1 = wmma_bf16(a, b1, acc1);
    }
    const int n0 = tileN + ln;
    const int n1 = tileN + 16 + ln;
    const float bv0 = bias[n0];
    const float bv1 = bias[n1];
    if (Cf) {
#pragma unroll
        for (int r = 0; r < 8; ++r) {
            const size_t m = (size_t)(tileM + r + hi * 8) * Nc;
            Cf[m + n0] = acc0[r] + bv0;
            Cf[m + n1] = acc1[r] + bv1;
        }
    }
    if (Cb) {
#pragma unroll
        for (int r = 0; r < 8; ++r) {
            const size_t m = (size_t)(tileM + r + hi * 8) * Nc;
            Cb[m + n0] = f2bf(acc0[r] + bv0);
            Cb[m + n1] = f2bf(acc1[r] + bv1);
        }
    }
}

// ---------------------------------------------------------------------------
// Fused cross-attention for one (graph b, 16-row tile).  8 waves = 8 heads.
// scores (16x256) in registers -> masked softmax (shfl reductions within the
// 16-lane halves of the C layout) -> ds_add_f32 head-average in LDS -> P
// staged bf16 in LDS -> P @ V via WMMA -> ctx (bf16, feeds Wo GEMM).
// dyn LDS: 8*16*256 bf16 (64 KB) + 16*256 f32 (16 KB) = 80 KB.
// ---------------------------------------------------------------------------
__global__ void attention_kernel(const __bf16* __restrict__ qh, const __bf16* __restrict__ kh,
                                 const __bf16* __restrict__ vhb, // [b][s/32][n(512)][s%32]
                                 const unsigned char* __restrict__ mask, // jnp bool_ = 1B
                                 __bf16* __restrict__ ctx, float* __restrict__ av_out) {
    extern __shared__ char smem_raw[];
    __bf16* pbuf = (__bf16*)smem_raw;                               // [H][16][256]
    float*  av   = (float*)(smem_raw + (size_t)H_N * 16 * 256 * 2); // [16][256]

    const int lane = threadIdx.x & 31;
    const int h    = threadIdx.x >> 5;
    const int hi   = lane >> 4;
    const int ln   = lane & 15;
    const int b    = blockIdx.x / (LMAX / 16);
    const int m0   = (blockIdx.x % (LMAX / 16)) * 16;

    for (int i = threadIdx.x; i < 16 * 256; i += blockDim.x) av[i] = 0.f;
    __syncthreads();

    // Q fragments for this head (dh=64 -> two K=32 chunks), loaded once.
    const __bf16* qrow = qh + (size_t)(b * LMAX + m0 + ln) * D_DIM + h * DH_DIM + hi * 16;
    v16bf aq0 = load_a_frag_bf(qrow);
    v16bf aq1 = load_a_frag_bf(qrow + 32);

    const unsigned char* mrow = mask + b * S_LEN;

    // scores = (Q K^T) / sqrt(64), masked.  K^T B-frags are contiguous along
    // features of kh (k runs along the feature dim).
    v8f sc[16];
#pragma unroll
    for (int j = 0; j < 16; ++j) {
        v8f acc = {};
        const __bf16* kbase = kh + (size_t)(b * S_LEN + j * 16 + ln) * D_DIM
                                 + h * DH_DIM + hi * 16;
        acc = wmma_bf16(aq0, load_b16(kbase), acc);
        acc = wmma_bf16(aq1, load_b16(kbase + 32), acc);
        const float madd = mrow[j * 16 + ln] ? -__builtin_inff() : 0.0f;
#pragma unroll
        for (int r = 0; r < 8; ++r) acc[r] = acc[r] * 0.125f + madd;
        sc[j] = acc;
    }

    // row-wise softmax: each row lives in the 16 lanes of one half-wave
    float mx[8], sm[8];
#pragma unroll
    for (int r = 0; r < 8; ++r) {
        float m = -__builtin_inff();
#pragma unroll
        for (int j = 0; j < 16; ++j) m = fmaxf(m, sc[j][r]);
#pragma unroll
        for (int o = 1; o <= 8; o <<= 1) m = fmaxf(m, __shfl_xor(m, o, 32));
        mx[r] = m;
    }
#pragma unroll
    for (int r = 0; r < 8; ++r) {
        float s = 0.f;
#pragma unroll
        for (int j = 0; j < 16; ++j) {
            float e = __expf(sc[j][r] - mx[r]);
            sc[j][r] = e;
            s += e;
        }
#pragma unroll
        for (int o = 1; o <= 8; o <<= 1) s += __shfl_xor(s, o, 32);
        sm[r] = 1.0f / s;
    }

    // normalize, head-average into LDS, stage bf16 P for the P@V WMMA
#pragma unroll
    for (int j = 0; j < 16; ++j) {
#pragma unroll
        for (int r = 0; r < 8; ++r) {
            float p = sc[j][r] * sm[r];
            int row = r + hi * 8;
            atomicAdd(&av[row * 256 + j * 16 + ln], p * (1.0f / H_N));
            pbuf[((h * 16 + row) * 256) + j * 16 + ln] = f2bf(p);
        }
    }

    // ctx[:, h*64 : h*64+64] = P @ V_h ; V in interleaved B layout
    const __bf16* prow = pbuf + (size_t)(h * 16 + ln) * 256 + hi * 16;
#pragma unroll
    for (int t = 0; t < 4; ++t) {
        v8f acc = {};
#pragma unroll
        for (int kk = 0; kk < 256; kk += 32) {
            v16bf a = load_a_frag_lds(prow + kk);
            const __bf16* vp = vhb + ((size_t)(b * 8 + (kk >> 5)) * D_DIM
                                      + h * DH_DIM + t * 16 + ln) * 32 + hi * 16;
            acc = wmma_bf16(a, load_b16(vp), acc);
        }
        const int n = h * DH_DIM + t * 16 + ln;
#pragma unroll
        for (int r = 0; r < 8; ++r)
            ctx[(size_t)(b * LMAX + m0 + r + hi * 8) * D_DIM + n] = f2bf(acc[r]);
    }

    __syncthreads();
    float* avdst = av_out + ((size_t)b * LMAX + m0) * S_LEN;
    for (int i = threadIdx.x; i < 16 * 256; i += blockDim.x) avdst[i] = av[i];
}

// ---------------------------------------------------------------------------
// out = LN(a + b); optional bf16 copy for the next WMMA consumer.
// warp per 512-col row; jnp.var is population variance, eps 1e-5.
// ---------------------------------------------------------------------------
__global__ void add_ln_kernel(const float* __restrict__ xa, const float* __restrict__ xb,
                              const float* __restrict__ g, const float* __restrict__ bt,
                              float* __restrict__ out, __bf16* __restrict__ outb, int rows) {
    const int warp = threadIdx.x >> 5, lane = threadIdx.x & 31;
    const int row = blockIdx.x * 8 + warp;
    if (row >= rows) return;
    const float* pa = xa + (size_t)row * D_DIM;
    const float* pb = xb + (size_t)row * D_DIM;
    float vals[16], s = 0.f, sq = 0.f;
#pragma unroll
    for (int i = 0; i < 16; ++i) {
        float v = pa[lane + i * 32] + pb[lane + i * 32];
        vals[i] = v; s += v; sq += v * v;
    }
#pragma unroll
    for (int o = 1; o < 32; o <<= 1) { s += __shfl_xor(s, o, 32); sq += __shfl_xor(sq, o, 32); }
    const float mean = s * (1.f / D_DIM);
    const float rstd = rsqrtf(sq * (1.f / D_DIM) - mean * mean + 1e-5f);
    float* po = out + (size_t)row * D_DIM;
    __bf16* pob = outb ? outb + (size_t)row * D_DIM : (__bf16*)0;
#pragma unroll
    for (int i = 0; i < 16; ++i) {
        int c = lane + i * 32;
        float r = (vals[i] - mean) * rstd * g[c] + bt[c];
        po[c] = r;
        if (pob) pob[c] = f2bf(r);
    }
}

// out = LN(leaky_relu(y) + x)  (FFN epilogue, slope 0.01)
__global__ void ffn_ln_kernel(const float* __restrict__ y, const float* __restrict__ x,
                              const float* __restrict__ g, const float* __restrict__ bt,
                              float* __restrict__ out, int rows) {
    const int warp = threadIdx.x >> 5, lane = threadIdx.x & 31;
    const int row = blockIdx.x * 8 + warp;
    if (row >= rows) return;
    const float* py = y + (size_t)row * D_DIM;
    const float* px = x + (size_t)row * D_DIM;
    float vals[16], s = 0.f, sq = 0.f;
#pragma unroll
    for (int i = 0; i < 16; ++i) {
        float yv = py[lane + i * 32];
        float v  = (yv > 0.f ? yv : 0.01f * yv) + px[lane + i * 32];
        vals[i] = v; s += v; sq += v * v;
    }
#pragma unroll
    for (int o = 1; o < 32; o <<= 1) { s += __shfl_xor(s, o, 32); sq += __shfl_xor(sq, o, 32); }
    const float mean = s * (1.f / D_DIM);
    const float rstd = rsqrtf(sq * (1.f / D_DIM) - mean * mean + 1e-5f);
    float* po = out + (size_t)row * D_DIM;
#pragma unroll
    for (int i = 0; i < 16; ++i) {
        int c = lane + i * 32;
        po[c] = (vals[i] - mean) * rstd * g[c] + bt[c];
    }
}

// ---------------------------------------------------------------------------
// conversion / layout-transform kernels (one pass each, off the critical path)
// ---------------------------------------------------------------------------
__global__ void f32_to_bf16_kernel(const float* __restrict__ src, __bf16* __restrict__ dst, long n) {
    long i = (long)blockIdx.x * blockDim.x + threadIdx.x;
    const long stride = (long)gridDim.x * blockDim.x;
    for (; i < n; i += stride) dst[i] = f2bf(src[i]);
}

// Wb[k/32][n][k%32] = W[k][n]  -> contiguous WMMA B-fragments
__global__ void convert_weight_kernel(const float* __restrict__ W, __bf16* __restrict__ Wb,
                                      int K, int Nc) {
    long i = (long)blockIdx.x * blockDim.x + threadIdx.x;
    const long total = (long)K * Nc;
    const long stride = (long)gridDim.x * blockDim.x;
    for (; i < total; i += stride) {
        int k = (int)(i / Nc), n = (int)(i % Nc);
        Wb[((size_t)(k >> 5) * Nc + n) * 32 + (k & 31)] = f2bf(W[i]);
    }
}

// vhb[b][s/32][n][s%32] = vh[b][s][n]  (bf16 -> bf16 relayout)
__global__ void transform_v_kernel(const __bf16* __restrict__ src, __bf16* __restrict__ dst) {
    long i = (long)blockIdx.x * blockDim.x + threadIdx.x;
    const long total = (long)B_G * S_LEN * D_DIM;
    const long stride = (long)gridDim.x * blockDim.x;
    for (; i < total; i += stride) {
        int n = (int)(i % D_DIM);
        long sa = i / D_DIM;
        int s = (int)(sa % S_LEN);
        int b = (int)(sa / S_LEN);
        dst[((size_t)(b * 8 + (s >> 5)) * D_DIM + n) * 32 + (s & 31)] = src[i];
    }
}

// ---------------------------------------------------------------------------
// ragged bookkeeping + pad/unpad
// ---------------------------------------------------------------------------
__global__ void compute_starts_kernel(const int* __restrict__ gb, int N, int* __restrict__ starts) {
    __shared__ int cnt[B_G];
    if (threadIdx.x < B_G) cnt[threadIdx.x] = 0;
    __syncthreads();
    for (int i = threadIdx.x; i < N; i += blockDim.x) atomicAdd(&cnt[gb[i]], 1);
    __syncthreads();
    if (threadIdx.x == 0) {
        int acc = 0;
        for (int b = 0; b < B_G; ++b) { starts[b] = acc; acc += cnt[b]; }
    }
}

__global__ void zero_f32_kernel(float* __restrict__ p, long n) {
    long i = (long)blockIdx.x * blockDim.x + threadIdx.x;
    const long stride = (long)gridDim.x * blockDim.x;
    for (; i < n; i += stride) p[i] = 0.f;
}

__global__ void scatter_pad_kernel(const float* __restrict__ nodes, const int* __restrict__ gb,
                                   const int* __restrict__ starts,
                                   float* __restrict__ dense, __bf16* __restrict__ denseb) {
    const int i = blockIdx.x;
    const int b = gb[i];
    const int p = i - starts[b];
    const float* src = nodes + (size_t)i * D_DIM;
    const size_t ro = ((size_t)b * LMAX + p) * D_DIM;
    for (int c = threadIdx.x; c < D_DIM; c += blockDim.x) {
        float v = src[c];
        dense[ro + c] = v;
        denseb[ro + c] = f2bf(v);
    }
}

__global__ void gather_unpad_kernel(const float* __restrict__ dense, const int* __restrict__ gb,
                                    const int* __restrict__ starts, float* __restrict__ out) {
    const int i = blockIdx.x;
    const int b = gb[i];
    const int p = i - starts[b];
    const float* src = dense + ((size_t)b * LMAX + p) * D_DIM;
    float* dst = out + (size_t)i * D_DIM;
    for (int c = threadIdx.x; c < D_DIM; c += blockDim.x) dst[c] = src[c];
}

// ---------------------------------------------------------------------------
extern "C" void kernel_launch(void* const* d_in, const int* in_sizes, int n_in,
                              void* d_out, int out_size, void* d_ws, size_t ws_size,
                              hipStream_t stream) {
    const float* graph_nodes = (const float*)d_in[0];
    const int*   graph_batch = (const int*)d_in[1];
    const float* cond        = (const float*)d_in[2];
    const unsigned char* cmask = (const unsigned char*)d_in[3]; // jnp bool_ (1B)
    // d_in[4]=max_nodes, d_in[5]=num_heads (compile-time constants here)
    const float* Wq   = (const float*)d_in[6];
    const float* bq   = (const float*)d_in[7];
    const float* Wk   = (const float*)d_in[8];
    const float* bk   = (const float*)d_in[9];
    const float* Wv   = (const float*)d_in[10];
    const float* bv   = (const float*)d_in[11];
    const float* in_w = (const float*)d_in[12]; // [3,512,512]
    const float* in_b = (const float*)d_in[13]; // [3,512]
    const float* Wo   = (const float*)d_in[14];
    const float* bo   = (const float*)d_in[15];
    const float* ln1g = (const float*)d_in[16];
    const float* ln1b = (const float*)d_in[17];
    const float* W1   = (const float*)d_in[18];
    const float* b1   = (const float*)d_in[19];
    const float* ln2g = (const float*)d_in[20];
    const float* ln2b = (const float*)d_in[21];

    const int N  = in_sizes[0] / D_DIM;  // 12032
    const int Mn = M_NODES;              // 15872
    const int Mc = B_G * S_LEN;          // 4096

    // ---------------- workspace partition ----------------
    const size_t BIG = (size_t)Mn * D_DIM;     // 8,126,464
    const size_t SML = (size_t)Mc * D_DIM;     // 2,097,152
    const size_t CND = (size_t)Mc * LAT_DIM;   // 3,145,728
    float* F0 = (float*)d_ws;          // dense -> y
    float* F1 = F0 + BIG;              // attn_out -> x2
    float* F2 = F1 + BIG;              // x1
    __bf16* G0    = (__bf16*)(F2 + BIG); // dense_b -> qh_b -> x1_b
    __bf16* G1    = G0 + BIG;            // q_b -> ctx_b
    __bf16* condb = G1 + BIG;
    __bf16* kbuf  = condb + CND;         // k_b -> vh(plain)
    __bf16* vbuf  = kbuf + SML;          // v_b -> vhb (interleaved)
    __bf16* khbuf = vbuf + SML;
    __bf16* wqb  = khbuf + SML;
    __bf16* wkb  = wqb  + 512 * 512;
    __bf16* wvb  = wkb  + 768 * 512;
    __bf16* iw0b = wvb  + 768 * 512;
    __bf16* iw1b = iw0b + 512 * 512;
    __bf16* iw2b = iw1b + 512 * 512;
    __bf16* wob  = iw2b + 512 * 512;
    __bf16* w1b  = wob  + 512 * 512;
    int* starts  = (int*)(w1b + 512 * 512);

    float* node_out = (float*)d_out;
    float* av_out   = node_out + (size_t)N * D_DIM;

    const dim3 blk256(256);

    // 0) ragged offsets + zero-padded dense (f32 + bf16 mirror)
    compute_starts_kernel<<<1, 256, 0, stream>>>(graph_batch, N, starts);
    zero_f32_kernel<<<2048, blk256, 0, stream>>>(F0, (long)BIG);
    zero_f32_kernel<<<1024, blk256, 0, stream>>>((float*)G0, (long)(BIG / 2));
    scatter_pad_kernel<<<N, blk256, 0, stream>>>(graph_nodes, graph_batch, starts, F0, G0);

    // 1) one-time bf16 conversions / B-layout transforms (cheap vs GEMMs)
    convert_weight_kernel<<<1024, blk256, 0, stream>>>(Wq, wqb, 512, 512);
    convert_weight_kernel<<<1536, blk256, 0, stream>>>(Wk, wkb, 768, 512);
    convert_weight_kernel<<<1536, blk256, 0, stream>>>(Wv, wvb, 768, 512);
    convert_weight_kernel<<<1024, blk256, 0, stream>>>(in_w,                 iw0b, 512, 512);
    convert_weight_kernel<<<1024, blk256, 0, stream>>>(in_w + 1 * 512 * 512, iw1b, 512, 512);
    convert_weight_kernel<<<1024, blk256, 0, stream>>>(in_w + 2 * 512 * 512, iw2b, 512, 512);
    convert_weight_kernel<<<1024, blk256, 0, stream>>>(Wo, wob, 512, 512);
    convert_weight_kernel<<<1024, blk256, 0, stream>>>(W1, w1b, 512, 512);
    f32_to_bf16_kernel<<<4096, blk256, 0, stream>>>(cond, condb, (long)CND);

    // 2) projections (bf16 WMMA, f32 accumulate; bf16-only outputs feed WMMA)
    gemm_bf16_kernel<<<dim3(2, Mn / 16), blk256, 0, stream>>>(G0, wqb, bq, nullptr, G1, Mn, D_DIM, D_DIM);                 // q
    gemm_bf16_kernel<<<dim3(2, Mc / 16), blk256, 0, stream>>>(condb, wkb, bk, nullptr, kbuf, Mc, D_DIM, LAT_DIM);          // k
    gemm_bf16_kernel<<<dim3(2, Mc / 16), blk256, 0, stream>>>(condb, wvb, bv, nullptr, vbuf, Mc, D_DIM, LAT_DIM);          // v
    gemm_bf16_kernel<<<dim3(2, Mn / 16), blk256, 0, stream>>>(G1, iw0b, in_b,           nullptr, G0, Mn, D_DIM, D_DIM);    // qh
    gemm_bf16_kernel<<<dim3(2, Mc / 16), blk256, 0, stream>>>(kbuf, iw1b, in_b + 512,   nullptr, khbuf, Mc, D_DIM, D_DIM); // kh
    gemm_bf16_kernel<<<dim3(2, Mc / 16), blk256, 0, stream>>>(vbuf, iw2b, in_b + 1024,  nullptr, kbuf, Mc, D_DIM, D_DIM);  // vh (plain, into kbuf)
    transform_v_kernel<<<4096, blk256, 0, stream>>>(kbuf, vbuf);  // vh -> interleaved B layout

    // 3) fused attention: qh(G0), kh, vhb -> ctx bf16 (G1) + av_attn (d_out)
    const size_t attn_lds = (size_t)H_N * 16 * 256 * 2 + (size_t)16 * 256 * 4; // 80 KB
    attention_kernel<<<B_G * (LMAX / 16), blk256, attn_lds, stream>>>(
        G0, khbuf, vbuf, cmask, G1, av_out);

    // 4) output projection + residual/LN + FFN + LN
    gemm_bf16_kernel<<<dim3(2, Mn / 16), blk256, 0, stream>>>(G1, wob, bo, F1, nullptr, Mn, D_DIM, D_DIM); // attn_out
    add_ln_kernel<<<Mn / 8, blk256, 0, stream>>>(F1, F0, ln1g, ln1b, F2, G0, Mn);                          // x1 (f32+bf16)
    gemm_bf16_kernel<<<dim3(2, Mn / 16), blk256, 0, stream>>>(G0, w1b, b1, F0, nullptr, Mn, D_DIM, D_DIM); // y
    ffn_ln_kernel<<<Mn / 8, blk256, 0, stream>>>(F0, F2, ln2g, ln2b, F1, Mn);                              // x2

    // 5) gather ragged node_out
    gather_unpad_kernel<<<N, blk256, 0, stream>>>(F1, graph_batch, starts, node_out);
}
